// RelativisticSelfAttention_45183055954006
// MI455X (gfx1250) — matrix operand order repaired and checked by reading, not compiled, not run on hardware
//
#include <hip/hip_runtime.h>
#include <hip/hip_bf16.h>

#define S_LEN 2048
#define D_DIM 1024
#define NH    16
#define HD    64

typedef __attribute__((ext_vector_type(16))) __bf16 bf16x16;
typedef __attribute__((ext_vector_type(8)))  float  f32x8;

struct Frag { uint4 lo, hi; };

__device__ __forceinline__ bf16x16 as_bf(const Frag& f) {
  return __builtin_bit_cast(bf16x16, f);
}

__device__ __forceinline__ unsigned short f2bf(float f) {
  unsigned int u = __float_as_uint(f);
  u += 0x7fffu + ((u >> 16) & 1u);       // round-to-nearest-even
  return (unsigned short)(u >> 16);
}
__device__ __forceinline__ float bf2f(unsigned short h) {
  return __uint_as_float(((unsigned int)h) << 16);
}
__device__ __forceinline__ unsigned int pack2(float a, float b) {
  return (unsigned int)f2bf(a) | ((unsigned int)f2bf(b) << 16);
}

// A-fragment: 16x32 bf16 tile at (row0, k0) of row-major [*, cols] (dword-packed).
// Lane m = lane&15 is the M row; per ISA: VGPR j<4 holds K=2j+8*half, j>=4 holds K=16+2(j-4)+8*half.
__device__ __forceinline__ Frag load_a_frag(const unsigned int* dw, int row_dw,
                                            int row0, int k0, int m, int half) {
  const uint4* p = (const uint4*)(dw + (size_t)(row0 + m) * row_dw + (k0 >> 1));
  Frag f; f.lo = p[half]; f.hi = p[2 + half]; return f;
}
// B-fragment: B(kk,n) = arr[col0+n][k0+kk].  Per ISA: VGPR j<4 holds K=2j+8*half, j>=4 holds K=16+2(j-4)+8*half at column n=lane&15.
__device__ __forceinline__ Frag load_b_frag(const unsigned int* dw, int row_dw,
                                            int col0, int k0, int n, int half) {
  const uint4* p = (const uint4*)(dw + (size_t)(col0 + n) * row_dw + (k0 >> 1));
  Frag f; f.lo = p[half]; f.hi = p[2 + half]; return f;
}

__device__ __forceinline__ f32x8 wmma_bf16(const Frag& a, const Frag& b, f32x8 c) {
  return __builtin_amdgcn_wmma_f32_16x16x32_bf16(false, as_bf(a), false, as_bf(b),
                                                 (short)0, c, false, false);
}

// ---------- prep: fp32 -> packed bf16 ----------
__global__ void pack_bf16_kernel(const float* __restrict__ src,
                                 unsigned int* __restrict__ dst, int n_pairs) {
  int stride = gridDim.x * blockDim.x;
  for (int i = blockIdx.x * blockDim.x + threadIdx.x; i < n_pairs; i += stride)
    dst[i] = pack2(src[2 * i], src[2 * i + 1]);
}

__global__ void init_ws_kernel(int* dmaxsq) {
  if (threadIdx.x == 0 && blockIdx.x == 0) *dmaxsq = 0;
}

// max squared pairwise distance (sqrt is monotone -> take sqrt later once)
__global__ void dmax_kernel(const float* __restrict__ pos, int* __restrict__ dmaxsq) {
  __shared__ float red[256];
  float mx = 0.f;
  long long total = (long long)S_LEN * S_LEN;
  for (long long p = (long long)blockIdx.x * blockDim.x + threadIdx.x; p < total;
       p += (long long)gridDim.x * blockDim.x) {
    int i = (int)(p >> 11);
    int j = (int)(p & (S_LEN - 1));
    float dx = pos[3 * i + 0] - pos[3 * j + 0];
    float dy = pos[3 * i + 1] - pos[3 * j + 1];
    float dz = pos[3 * i + 2] - pos[3 * j + 2];
    mx = fmaxf(mx, dx * dx + dy * dy + dz * dz);
  }
  red[threadIdx.x] = mx;
  __syncthreads();
  for (int o = 128; o > 0; o >>= 1) {
    if (threadIdx.x < o) red[threadIdx.x] = fmaxf(red[threadIdx.x], red[threadIdx.x + o]);
    __syncthreads();
  }
  if (threadIdx.x == 0) atomicMax(dmaxsq, __float_as_int(red[0]));
}

// ---------- Y = X @ W^T + b, computed "swapped" (C rows = out-feature) -> row-major bf16 Y[S][D]
__global__ void gemm_proj_bf16(const unsigned int* __restrict__ xbf,
                               const unsigned int* __restrict__ wbf,
                               const float* __restrict__ bias,
                               unsigned int* __restrict__ ybf) {
  int tid = blockIdx.x * blockDim.x + threadIdx.x;
  int wave = tid >> 5, lane = tid & 31;
  int half = lane >> 4, n = lane & 15;
  int ot = wave & 63, st = wave >> 6;          // 64 o-tiles x 32 s-strips
  int o0 = ot * 16, s0 = st * 64;
  f32x8 z = {0.f, 0.f, 0.f, 0.f, 0.f, 0.f, 0.f, 0.f};
  f32x8 acc[4] = {z, z, z, z};
  for (int k0 = 0; k0 < D_DIM; k0 += 32) {
    Frag a = load_a_frag(wbf, D_DIM / 2, o0, k0, n, half);   // A = W rows (o)
#pragma unroll
    for (int t = 0; t < 4; ++t) {
      Frag b = load_b_frag(xbf, D_DIM / 2, s0 + t * 16, k0, n, half); // B(kk=d,n=s)=x[s][d]
      acc[t] = wmma_bf16(a, b, acc[t]);
    }
  }
  float bv[8];
#pragma unroll
  for (int v = 0; v < 8; ++v) bv[v] = bias[o0 + 8 * half + v];
#pragma unroll
  for (int t = 0; t < 4; ++t) {
    int s = s0 + t * 16 + n;
    unsigned int* dst = ybf + (size_t)s * (D_DIM / 2) + ((o0 + 8 * half) >> 1);
    uint4 w;
    w.x = pack2(acc[t][0] + bv[0], acc[t][1] + bv[1]);
    w.y = pack2(acc[t][2] + bv[2], acc[t][3] + bv[3]);
    w.z = pack2(acc[t][4] + bv[4], acc[t][5] + bv[5]);
    w.w = pack2(acc[t][6] + bv[6], acc[t][7] + bv[7]);
    *(uint4*)dst = w;
  }
}

// ---------- V projection (normal orientation), stored transposed: Vt[d][s] bf16
__global__ void gemm_vt_bf16(const unsigned int* __restrict__ xbf,
                             const unsigned int* __restrict__ wbf,
                             const float* __restrict__ bias,
                             unsigned int* __restrict__ vt) {
  int tid = blockIdx.x * blockDim.x + threadIdx.x;
  int wave = tid >> 5, lane = tid & 31;
  int half = lane >> 4, n = lane & 15;
  int st = wave & 127, os = wave >> 7;          // 128 s-tiles x 16 o-strips
  int s0 = st * 16, ob = os * 64;
  f32x8 z = {0.f, 0.f, 0.f, 0.f, 0.f, 0.f, 0.f, 0.f};
  f32x8 acc[4] = {z, z, z, z};
  for (int k0 = 0; k0 < D_DIM; k0 += 32) {
    Frag a = load_a_frag(xbf, D_DIM / 2, s0, k0, n, half);   // A = x rows (s)
#pragma unroll
    for (int t = 0; t < 4; ++t) {
      Frag b = load_b_frag(wbf, D_DIM / 2, ob + t * 16, k0, n, half); // B(kk=d,n=o)=W[o][d]
      acc[t] = wmma_bf16(a, b, acc[t]);
    }
  }
#pragma unroll
  for (int t = 0; t < 4; ++t) {
    float bb = bias[ob + t * 16 + n];
    unsigned int* dst = vt + (size_t)(ob + t * 16 + n) * (S_LEN / 2) + ((s0 + 8 * half) >> 1);
    uint4 w;
    w.x = pack2(acc[t][0] + bb, acc[t][1] + bb);
    w.y = pack2(acc[t][2] + bb, acc[t][3] + bb);
    w.z = pack2(acc[t][4] + bb, acc[t][5] + bb);
    w.w = pack2(acc[t][6] + bb, acc[t][7] + bb);
    *(uint4*)dst = w;
  }
}

// ---------- output projection: out = O @ Wo^T + bo, fp32 row-major
__global__ void gemm_out_f32(const unsigned int* __restrict__ obf,
                             const unsigned int* __restrict__ wbf,
                             const float* __restrict__ bias,
                             float* __restrict__ out) {
  int tid = blockIdx.x * blockDim.x + threadIdx.x;
  int wave = tid >> 5, lane = tid & 31;
  int half = lane >> 4, n = lane & 15;
  int ot = wave & 63, st = wave >> 6;
  int o0 = ot * 16, s0 = st * 64;
  f32x8 z = {0.f, 0.f, 0.f, 0.f, 0.f, 0.f, 0.f, 0.f};
  f32x8 acc[4] = {z, z, z, z};
  for (int k0 = 0; k0 < D_DIM; k0 += 32) {
    Frag a = load_a_frag(wbf, D_DIM / 2, o0, k0, n, half);
#pragma unroll
    for (int t = 0; t < 4; ++t) {
      Frag b = load_b_frag(obf, D_DIM / 2, s0 + t * 16, k0, n, half);
      acc[t] = wmma_bf16(a, b, acc[t]);
    }
  }
  float bv[8];
#pragma unroll
  for (int v = 0; v < 8; ++v) bv[v] = bias[o0 + 8 * half + v];
#pragma unroll
  for (int t = 0; t < 4; ++t) {
    int s = s0 + t * 16 + n;
    float* dst = out + (size_t)s * D_DIM + o0 + 8 * half;
    float4 lo = {acc[t][0] + bv[0], acc[t][1] + bv[1], acc[t][2] + bv[2], acc[t][3] + bv[3]};
    float4 hi = {acc[t][4] + bv[4], acc[t][5] + bv[5], acc[t][6] + bv[6], acc[t][7] + bv[7]};
    *(float4*)dst = lo;
    *(float4*)(dst + 4) = hi;
  }
}

// ---------- attention: per (head, 16-query block): QK^T -> modulate -> softmax -> PV
__global__ void __launch_bounds__(32)
attn_kernel(const unsigned int* __restrict__ qbf, const unsigned int* __restrict__ kbf,
            const unsigned int* __restrict__ vt, const float* __restrict__ positions,
            const int* __restrict__ amask, const float* __restrict__ velocity,
            const int* __restrict__ dmaxsq_bits, unsigned int* __restrict__ obf) {
  __shared__ alignas(16) unsigned int sc[16 * (S_LEN / 2)];  // 16 rows x 2048 bf16 = 64KB
  unsigned short* scs = (unsigned short*)sc;

  int lane = threadIdx.x;
  int half = lane >> 4, n = lane & 15;
  int h = blockIdx.x >> 7;         // S/16 = 128 q-blocks per head
  int q0 = (blockIdx.x & 127) * 16;

  float vh = fminf(fabsf(velocity[h]), 0.999f);
  float gamma = rsqrtf(1.0f - vh * vh);
  float coef = gamma * vh;
  float dmax = sqrtf(__int_as_float(*dmaxsq_bits));
  float dinv = dmax > 0.f ? 1.0f / (dmax + 1e-8f) : 0.f;
  const float scale = 0.125f;      // 64^-0.5

  // positions of this lane's 8 query rows (C rows m = v + 8*half)
  float qx[8], qy[8], qz[8];
#pragma unroll
  for (int v = 0; v < 8; ++v) {
    int q = q0 + 8 * half + v;
    qx[v] = positions[3 * q + 0];
    qy[v] = positions[3 * q + 1];
    qz[v] = positions[3 * q + 2];
  }
  Frag qa0 = load_a_frag(qbf, D_DIM / 2, q0, h * HD + 0, n, half);
  Frag qa1 = load_a_frag(qbf, D_DIM / 2, q0, h * HD + 32, n, half);

  // Phase 1: scaled, distance-modulated, masked scores -> LDS (bf16)
  f32x8 z = {0.f, 0.f, 0.f, 0.f, 0.f, 0.f, 0.f, 0.f};
  for (int kb = 0; kb < S_LEN / 16; ++kb) {
    int key = kb * 16 + n;
    Frag b0 = load_b_frag(kbf, D_DIM / 2, kb * 16, h * HD + 0, n, half);
    Frag b1 = load_b_frag(kbf, D_DIM / 2, kb * 16, h * HD + 32, n, half);
    f32x8 acc = z;
    acc = wmma_bf16(qa0, b0, acc);
    acc = wmma_bf16(qa1, b1, acc);
    float px = positions[3 * key + 0], py = positions[3 * key + 1], pz = positions[3 * key + 2];
    int mk = amask[key];
#pragma unroll
    for (int v = 0; v < 8; ++v) {
      float dx = qx[v] - px, dy = qy[v] - py, dz = qz[v] - pz;
      float dist = sqrtf(dx * dx + dy * dy + dz * dz) * dinv;
      float sv = acc[v] * scale * __expf(-dist * coef);
      if (!mk) sv = -1e30f;
      scs[(8 * half + v) * S_LEN + key] = f2bf(sv);
    }
  }
  __syncthreads();

  // Phase 2: two-pass softmax per row; keep 1/rowsum for lane's own column row
  float myinv = 1.f;
  for (int m = 0; m < 16; ++m) {
    float mx = -3.0e38f;
    for (int j = lane; j < S_LEN; j += 32) mx = fmaxf(mx, bf2f(scs[m * S_LEN + j]));
    for (int off = 16; off; off >>= 1) mx = fmaxf(mx, __shfl_xor(mx, off, 32));
    float sum = 0.f;
    for (int j = lane; j < S_LEN; j += 32) {
      float e = __expf(bf2f(scs[m * S_LEN + j]) - mx);
      sum += e;
      scs[m * S_LEN + j] = f2bf(e);   // unnormalized probs back to LDS
    }
    for (int off = 16; off; off >>= 1) sum += __shfl_xor(sum, off, 32);
    if (n == m) myinv = 1.0f / sum;
  }
  __syncthreads();

  // Phase 3: O^T[d][q] = sum_k Vt[d][k] * P[q][k]  (A = Vt row-major, B = P from LDS)
  f32x8 oacc[4] = {z, z, z, z};
  for (int k0 = 0; k0 < S_LEN; k0 += 32) {
    const uint4* pB = (const uint4*)(sc + (size_t)n * (S_LEN / 2) + (k0 >> 1));
    Frag b; b.lo = pB[half]; b.hi = pB[2 + half];
#pragma unroll
    for (int t = 0; t < 4; ++t) {
      Frag a = load_a_frag(vt, S_LEN / 2, h * HD + t * 16, k0, n, half);
      oacc[t] = wmma_bf16(a, b, oacc[t]);
    }
  }
  // store O[q][d] bf16 row-major; fold 1/rowsum (lane's q = q0+n)
#pragma unroll
  for (int t = 0; t < 4; ++t) {
    unsigned int* dst = obf + (size_t)(q0 + n) * (D_DIM / 2) + ((h * HD + t * 16 + 8 * half) >> 1);
    uint4 w;
    w.x = pack2(oacc[t][0] * myinv, oacc[t][1] * myinv);
    w.y = pack2(oacc[t][2] * myinv, oacc[t][3] * myinv);
    w.z = pack2(oacc[t][4] * myinv, oacc[t][5] * myinv);
    w.w = pack2(oacc[t][6] * myinv, oacc[t][7] * myinv);
    *(uint4*)dst = w;
  }
}

extern "C" void kernel_launch(void* const* d_in, const int* in_sizes, int n_in,
                              void* d_out, int out_size, void* d_ws, size_t ws_size,
                              hipStream_t stream) {
  (void)in_sizes; (void)n_in; (void)out_size; (void)ws_size;
  const float* x    = (const float*)d_in[0];
  const int*   am   = (const int*)d_in[1];
  const float* pos  = (const float*)d_in[2];
  const float* Wq   = (const float*)d_in[3];
  const float* bq   = (const float*)d_in[4];
  const float* Wk   = (const float*)d_in[5];
  const float* bk   = (const float*)d_in[6];
  const float* Wv   = (const float*)d_in[7];
  const float* bv   = (const float*)d_in[8];
  const float* Wo   = (const float*)d_in[9];
  const float* bo   = (const float*)d_in[10];
  const float* vel  = (const float*)d_in[11];
  float* out = (float*)d_out;

  char* ws = (char*)d_ws;
  size_t off = 0;
  auto bump = [&](size_t bytes) -> void* {
    void* p = ws + off;
    off = (off + bytes + 255) & ~(size_t)255;
    return p;
  };
  int*          dmaxsq = (int*)bump(4);
  unsigned int* xbf = (unsigned int*)bump((size_t)S_LEN * D_DIM * 2);
  unsigned int* qb  = (unsigned int*)bump((size_t)S_LEN * D_DIM * 2);
  unsigned int* kb2 = (unsigned int*)bump((size_t)S_LEN * D_DIM * 2);
  unsigned int* vtb = (unsigned int*)bump((size_t)S_LEN * D_DIM * 2);
  unsigned int* ob  = (unsigned int*)bump((size_t)S_LEN * D_DIM * 2);
  unsigned int* wqb = (unsigned int*)bump((size_t)D_DIM * D_DIM * 2);
  unsigned int* wkb = (unsigned int*)bump((size_t)D_DIM * D_DIM * 2);
  unsigned int* wvb = (unsigned int*)bump((size_t)D_DIM * D_DIM * 2);
  unsigned int* wob = (unsigned int*)bump((size_t)D_DIM * D_DIM * 2);

  const int npx = S_LEN * D_DIM / 2;   // x pairs
  const int npw = D_DIM * D_DIM / 2;   // weight pairs
  pack_bf16_kernel<<<1024, 256, 0, stream>>>(x,  xbf, npx);
  pack_bf16_kernel<<<1024, 256, 0, stream>>>(Wq, wqb, npw);
  pack_bf16_kernel<<<1024, 256, 0, stream>>>(Wk, wkb, npw);
  pack_bf16_kernel<<<1024, 256, 0, stream>>>(Wv, wvb, npw);
  pack_bf16_kernel<<<1024, 256, 0, stream>>>(Wo, wob, npw);

  init_ws_kernel<<<1, 32, 0, stream>>>(dmaxsq);
  dmax_kernel<<<1024, 256, 0, stream>>>(pos, dmaxsq);

  gemm_proj_bf16<<<512, 128, 0, stream>>>(xbf, wqb, bq, qb);
  gemm_proj_bf16<<<512, 128, 0, stream>>>(xbf, wkb, bk, kb2);
  gemm_vt_bf16 <<<512, 128, 0, stream>>>(xbf, wvb, bv, vtb);

  attn_kernel<<<NH * (S_LEN / 16), 32, 0, stream>>>(qb, kb2, vtb, pos, am, vel, dmaxsq, ob);

  gemm_out_f32<<<512, 128, 0, stream>>>(ob, wob, bo, out);
}
